// LilaModel_62543313764834
// MI455X (gfx1250) — matrix operand order repaired and checked by reading, not compiled, not run on hardware
//
#include <hip/hip_runtime.h>

typedef _Float16 v16h __attribute__((ext_vector_type(16)));
typedef _Float16 v8h  __attribute__((ext_vector_type(8)));
typedef _Float16 v2h  __attribute__((ext_vector_type(2)));
typedef float    v8f  __attribute__((ext_vector_type(8)));

union V16 { v16h v; v8h h[2]; };

#define NPATCH 36
// ---- workspace layout (bytes) ----
#define OFF_W1   0
#define SZ_W1    (NPATCH*3*512*2)          // 108 B-tiles (K-chunks of 32) -> 110592 B
#define OFF_W2   (OFF_W1 + SZ_W1)          // 1 tile
#define OFF_W3   (OFF_W2 + 1024)           // 1 tile (only N==0 column nonzero)
#define OFF_E1   (OFF_W3 + 1024)           // 4 tiles (2 K-chunks x 2 N-halves)
#define OFF_E2   (OFF_E1 + 4096)           // 1 tile
#define OFF_E3   (OFF_E2 + 1024)           // 1 tile (only N==0 column nonzero)
#define OFF_BIAS (OFF_E3 + 1024)           // 82 floats: sb1[16] sb2[16] sb3 eb1[32] eb2[16] eb3
#define WS_BYTES (OFF_BIAS + 82*4)

// ---- shared memory layout (bytes); each wave owns 32 samples ----
#define SM_WEIGHTS OFF_BIAS                 // verbatim copy of all tiles (118784 B)
#define SM_INP     118784                   // 8 waves * 32 samples * 96 halves   (6144 B/wave)
#define SM_STAGE   (SM_INP + 8*6144)        // 8 waves * 32x32 halves             (2048 B/wave)
#define SM_SIGHT   (SM_STAGE + 8*2048)      // 8 waves * 32 samples * 64 halves   (4096 B/wave)
#define SM_TOTAL   (SM_SIGHT + 8*4096)      // 217088 B (< 320 KB WGP LDS)

#define V8F_ZERO {0.f,0.f,0.f,0.f,0.f,0.f,0.f,0.f}
#define WMMA(A,B,C) __builtin_amdgcn_wmma_f32_16x16x32_f16(false,(A),false,(B),(short)0,(C),false,false)

__device__ __forceinline__ float fast_tanh(float x) {
  float c = fminf(fmaxf(x, -9.0f), 9.0f);
  float e = __expf(2.0f * c);                       // v_exp_f32 (TRANS, co-executes with WMMA)
  return (e - 1.0f) * __builtin_amdgcn_rcpf(e + 1.0f);
}

// ------------------------------------------------------------------
// Prep kernel: fold patch-gather into W1, pre-swizzle all weights into
// the CDNA5 WMMA B-tile lane order:  tile[lane][t] = B[(lane>>4)*8+(t&7)+(t>>3)*16][lane&15]
// ------------------------------------------------------------------
__global__ __launch_bounds__(256) void lila_prep(
    const float* __restrict__ sW1, const float* __restrict__ sb1,
    const float* __restrict__ sW2, const float* __restrict__ sb2,
    const float* __restrict__ sW3, const float* __restrict__ sb3,
    const float* __restrict__ eW1, const float* __restrict__ eb1,
    const float* __restrict__ eW2, const float* __restrict__ eb2,
    const float* __restrict__ eW3, const float* __restrict__ eb3,
    char* __restrict__ ws)
{
  const int tid = threadIdx.x;
  _Float16* w1t = (_Float16*)(ws + OFF_W1);
  _Float16* w2t = (_Float16*)(ws + OFF_W2);
  _Float16* w3t = (_Float16*)(ws + OFF_W3);
  _Float16* e1t = (_Float16*)(ws + OFF_E1);
  _Float16* e2t = (_Float16*)(ws + OFF_E2);
  _Float16* e3t = (_Float16*)(ws + OFF_E3);
  float*    bst = (float*)(ws + OFF_BIAS);

  // Folded layer-1 weights: W1fold[p][i][o] = sum over patch-features mapping to input i.
  // Patch p=(x=p%6, y=p/6); feature f=(ox*3+oy)*6+j reads input index (x+ox)*8+(y+oy)+j (all < 69).
  for (int e = tid; e < NPATCH*3*512; e += 256) {
    int t = e & 15, lane = (e >> 4) & 31, tile = e >> 9;
    int c = tile % 3, p = tile / 3;
    int N = lane & 15;
    int K = c*32 + (lane >> 4)*8 + (t & 7) + (t >> 3)*16;
    float acc = 0.0f;
    if (K < 69) {
      int x = p % 6, y = p / 6;
      for (int ox = 0; ox < 3; ++ox)
        for (int oy = 0; oy < 3; ++oy)
          for (int j = 0; j < 6; ++j)
            if ((x + ox)*8 + (y + oy) + j == K)
              acc += sW1[((ox*3 + oy)*6 + j)*16 + N];
    }
    w1t[e] = (_Float16)acc;
  }
  // 16x16 matrices framed in K=32 tiles (upper K zero); N==0-only tiles for the 16->1 layers.
  for (int e = tid; e < 512; e += 256) {
    int t = e & 15, lane = e >> 4;
    int N = lane & 15, K = (lane >> 4)*8 + (t & 7) + (t >> 3)*16;
    w2t[e] = (_Float16)((K < 16) ? sW2[K*16 + N] : 0.0f);
    w3t[e] = (_Float16)((K < 16 && N == 0) ? sW3[K] : 0.0f);
    e2t[e] = (_Float16)eW2[K*16 + N];                       // 32x16: K always < 32
    e3t[e] = (_Float16)((K < 16 && N == 0) ? eW3[K] : 0.0f);
  }
  // eW1 (64x32): 2 K-chunks x 2 N-halves
  for (int e = tid; e < 4*512; e += 256) {
    int t = e & 15, lane = (e >> 4) & 31, tile = e >> 9;
    int c = tile >> 1, nh = tile & 1;
    int N = nh*16 + (lane & 15);
    int K = c*32 + (lane >> 4)*8 + (t & 7) + (t >> 3)*16;
    e1t[e] = (_Float16)eW1[K*32 + N];
  }
  if (tid < 16)       bst[tid] = sb1[tid];
  else if (tid < 32)  bst[tid] = sb2[tid - 16];
  else if (tid == 32) bst[32]  = sb3[0];
  else if (tid < 65)  bst[tid] = eb1[tid - 33];
  else if (tid < 81)  bst[tid] = eb2[tid - 65];
  else if (tid == 81) bst[81]  = eb3[0];
}

// ------------------------------------------------------------------
// Main kernel: 8 waves/block, 32 samples/wave (two 16-row M-tiles so every
// B-tile load feeds two independent WMMAs), grid 512 -> 131072 samples.
// ------------------------------------------------------------------
__global__ __launch_bounds__(256, 1) void lila_main(
    const float* __restrict__ inp, const char* __restrict__ ws,
    float* __restrict__ out)
{
  __shared__ __align__(16) char smem[SM_TOTAL];

  // Stage all swizzled weight tiles into LDS (shared by all 8 waves).
  {
    const uint4* src = (const uint4*)ws;
    uint4* dst = (uint4*)smem;
    for (int i = threadIdx.x; i < SM_WEIGHTS/16; i += 256) dst[i] = src[i];
  }
  __syncthreads();

  const _Float16* w1t = (const _Float16*)(smem + OFF_W1);
  const _Float16* w2t = (const _Float16*)(smem + OFF_W2);
  const _Float16* w3t = (const _Float16*)(smem + OFF_W3);
  const _Float16* e1t = (const _Float16*)(smem + OFF_E1);
  const _Float16* e2t = (const _Float16*)(smem + OFF_E2);
  const _Float16* e3t = (const _Float16*)(smem + OFF_E3);
  const float*    bst = (const float*)(ws + OFF_BIAS);

  const int wave = threadIdx.x >> 5;
  const int lane = threadIdx.x & 31;
  const int n = lane & 15;      // N / sample-row index within half-wave
  const int g = lane >> 4;      // half-wave group

  _Float16* myInp   = (_Float16*)(smem + SM_INP   + wave*6144);   // [32][96] f16
  _Float16* myStage = (_Float16*)(smem + SM_STAGE + wave*2048);   // [32][16] or [32][32] f16
  _Float16* mySight = (_Float16*)(smem + SM_SIGHT + wave*4096);   // [32][64] f16

  const float b1 = bst[n], b2 = bst[16 + n], b3 = bst[32];
  const float be1[2] = { bst[33 + n], bst[49 + n] };
  const float be2 = bst[65 + n], be3 = bst[81];

  const long sampleBase = (long)(blockIdx.x * 8 + wave) * 32;
  const float* ibase = inp + sampleBase * 384;

  // Load only features 0..68 of each sample (rest of the 384 are never indexed
  // by PATCH_IDX); convert to f16, pad to 96.
  for (int u = 0; u < 2; ++u) {
    const int s = u*16 + n;
    for (int c4 = g; c4 < 24; c4 += 2) {
      const int c = c4 * 4;
      float4 v = make_float4(0.f, 0.f, 0.f, 0.f);
      if (c < 69) v = *(const float4*)(ibase + (long)s * 384 + c);
      v2h h01 = { (_Float16)((c + 0 < 69) ? v.x : 0.f),
                  (_Float16)((c + 1 < 69) ? v.y : 0.f) };
      v2h h23 = { (_Float16)((c + 2 < 69) ? v.z : 0.f),
                  (_Float16)((c + 3 < 69) ? v.w : 0.f) };
      *(v2h*)(myInp + s*96 + c)     = h01;
      *(v2h*)(myInp + s*96 + c + 2) = h23;
    }
  }
  // zero-pad "vision" columns 36..63 for all 32 samples
  for (int i = lane; i < 32*28; i += 32)
    mySight[(i / 28)*64 + 36 + (i % 28)] = (_Float16)0.0f;
  __builtin_amdgcn_wave_barrier();     // same-wave LDS ops are in-order in HW

  // A-matrices (2 tiles x 16 samples x 96 features), shared by all 36 patches.
  // ISA A layout: lane holds M=lane&15, K = chunk + (lane>>4)*8 + (0..7) [v0-3] and +16 [v4-7].
  V16 a1[2][3];
#pragma unroll
  for (int u = 0; u < 2; ++u)
#pragma unroll
    for (int c = 0; c < 3; ++c) {
      const _Float16* ab = myInp + (u*16 + n)*96 + c*32 + g*8;
      a1[u][c].h[0] = *(const v8h*)(ab);
      a1[u][c].h[1] = *(const v8h*)(ab + 16);
    }

  V16 bW2, bW3;
  bW2.h[0] = *(const v8h*)(w2t + lane*16); bW2.h[1] = *(const v8h*)(w2t + lane*16 + 8);
  bW3.h[0] = *(const v8h*)(w3t + lane*16); bW3.h[1] = *(const v8h*)(w3t + lane*16 + 8);

  const v8h zero8 = { (_Float16)0.f,(_Float16)0.f,(_Float16)0.f,(_Float16)0.f,
                      (_Float16)0.f,(_Float16)0.f,(_Float16)0.f,(_Float16)0.f };

  // Per-patch B-tile loader (6 ds_load_b128 into a statically named register buffer).
  auto loadB = [&](int p, V16 (&dst)[3]) {
#pragma unroll
    for (int c = 0; c < 3; ++c) {
      const _Float16* bb = w1t + (p*3 + c)*512 + lane*16;
      dst[c].h[0] = *(const v8h*)(bb);
      dst[c].h[1] = *(const v8h*)(bb + 8);
    }
  };

  // Full per-patch sight pipeline: layer1 (3 WMMA x 2 M-tiles) -> layer2 -> layer3.
  auto doPatch = [&](int p, V16 (&bcur)[3]) {
    v8f accL[2] = { V8F_ZERO, V8F_ZERO };
#pragma unroll
    for (int c = 0; c < 3; ++c)
#pragma unroll
      for (int u = 0; u < 2; ++u)
        accL[u] = WMMA(a1[u][c].v, bcur[c].v, accL[u]);
#pragma unroll
    for (int u = 0; u < 2; ++u)
#pragma unroll
      for (int r = 0; r < 8; ++r)                  // C layout: M=r+g*8, N=n
        myStage[(u*16 + r + g*8)*16 + n] = (_Float16)fast_tanh(accL[u][r] + b1);
    __builtin_amdgcn_wave_barrier();

    // layer 2 (16->16, K padded to 32): two independent chains
    V16 a2[2];
    v8f acc2[2] = { V8F_ZERO, V8F_ZERO };
#pragma unroll
    for (int u = 0; u < 2; ++u) {
      a2[u].h[0] = *(const v8h*)(myStage + (u*16 + n)*16 + g*8);
      a2[u].h[1] = zero8;
    }
#pragma unroll
    for (int u = 0; u < 2; ++u) acc2[u] = WMMA(a2[u].v, bW2.v, acc2[u]);
#pragma unroll
    for (int u = 0; u < 2; ++u)
#pragma unroll
      for (int r = 0; r < 8; ++r)
        myStage[(u*16 + r + g*8)*16 + n] = (_Float16)fast_tanh(acc2[u][r] + b2);
    __builtin_amdgcn_wave_barrier();

    // layer 3 (16->1): B has only column 0 nonzero
    V16 a3[2];
    v8f acc3[2] = { V8F_ZERO, V8F_ZERO };
#pragma unroll
    for (int u = 0; u < 2; ++u) {
      a3[u].h[0] = *(const v8h*)(myStage + (u*16 + n)*16 + g*8);
      a3[u].h[1] = zero8;
    }
#pragma unroll
    for (int u = 0; u < 2; ++u) acc3[u] = WMMA(a3[u].v, bW3.v, acc3[u]);
    if (n == 0) {
#pragma unroll
      for (int u = 0; u < 2; ++u)
#pragma unroll
        for (int r = 0; r < 8; ++r)
          mySight[(u*16 + r + g*8)*64 + p] = (_Float16)fast_tanh(acc3[u][r] + b3);
    }
    __builtin_amdgcn_wave_barrier();
  };

  // Software-pipelined patch loop, unrolled by 2 with statically named double
  // buffers so the B tiles stay in VGPRs (no alloca->LDS demotion).
  V16 bt0[3], bt1[3];
  loadB(0, bt0);
  for (int p = 0; p < NPATCH; p += 2) {
    loadB(p + 1, bt1);               // prefetch odd patch while even computes
    doPatch(p, bt0);
    if (p + 2 < NPATCH) loadB(p + 2, bt0);
    doPatch(p + 1, bt1);
  }

  // ---- eye network: vision(64) -> 32 -> 16 -> 1, two M-tiles throughout ----
  V16 ae[2][2];
#pragma unroll
  for (int u = 0; u < 2; ++u)
#pragma unroll
    for (int c = 0; c < 2; ++c) {
      const _Float16* ab = mySight + (u*16 + n)*64 + c*32 + g*8;
      ae[u][c].h[0] = *(const v8h*)(ab);
      ae[u][c].h[1] = *(const v8h*)(ab + 16);
    }
#pragma unroll
  for (int nh = 0; nh < 2; ++nh) {                 // two 16-wide N tiles of eW1
    V16 be[2];
#pragma unroll
    for (int c = 0; c < 2; ++c) {
      const _Float16* bb = e1t + (c*2 + nh)*512 + lane*16;
      be[c].h[0] = *(const v8h*)(bb); be[c].h[1] = *(const v8h*)(bb + 8);
    }
    v8f acc[2] = { V8F_ZERO, V8F_ZERO };
#pragma unroll
    for (int c = 0; c < 2; ++c)
#pragma unroll
      for (int u = 0; u < 2; ++u)
        acc[u] = WMMA(ae[u][c].v, be[c].v, acc[u]);
#pragma unroll
    for (int u = 0; u < 2; ++u)
#pragma unroll
      for (int r = 0; r < 8; ++r)
        myStage[(u*16 + r + g*8)*32 + nh*16 + n] = (_Float16)fast_tanh(acc[u][r] + be1[nh]);
  }
  __builtin_amdgcn_wave_barrier();

  V16 bE2;
  bE2.h[0] = *(const v8h*)(e2t + lane*16); bE2.h[1] = *(const v8h*)(e2t + lane*16 + 8);
  V16 ag[2];
  v8f accg[2] = { V8F_ZERO, V8F_ZERO };
#pragma unroll
  for (int u = 0; u < 2; ++u) {
    ag[u].h[0] = *(const v8h*)(myStage + (u*16 + n)*32 + g*8);
    ag[u].h[1] = *(const v8h*)(myStage + (u*16 + n)*32 + 16 + g*8);
  }
#pragma unroll
  for (int u = 0; u < 2; ++u) accg[u] = WMMA(ag[u].v, bE2.v, accg[u]);
#pragma unroll
  for (int u = 0; u < 2; ++u)
#pragma unroll
    for (int r = 0; r < 8; ++r)
      myStage[(u*16 + r + g*8)*16 + n] = (_Float16)fast_tanh(accg[u][r] + be2);
  __builtin_amdgcn_wave_barrier();

  V16 bE3;
  bE3.h[0] = *(const v8h*)(e3t + lane*16); bE3.h[1] = *(const v8h*)(e3t + lane*16 + 8);
  V16 a4[2];
  v8f acco[2] = { V8F_ZERO, V8F_ZERO };
#pragma unroll
  for (int u = 0; u < 2; ++u) {
    a4[u].h[0] = *(const v8h*)(myStage + (u*16 + n)*16 + g*8);
    a4[u].h[1] = zero8;
  }
#pragma unroll
  for (int u = 0; u < 2; ++u) acco[u] = WMMA(a4[u].v, bE3.v, acco[u]);
  if (n == 0) {
#pragma unroll
    for (int u = 0; u < 2; ++u)
#pragma unroll
      for (int r = 0; r < 8; ++r)
        out[sampleBase + u*16 + r + g*8] = fast_tanh(acco[u][r] + be3);
  }
}

// ------------------------------------------------------------------
extern "C" void kernel_launch(void* const* d_in, const int* in_sizes, int n_in,
                              void* d_out, int out_size, void* d_ws, size_t ws_size,
                              hipStream_t stream)
{
  (void)in_sizes; (void)n_in; (void)out_size; (void)ws_size;
  const float* inp = (const float*)d_in[0];
  const float* sW1 = (const float*)d_in[1];
  const float* sb1 = (const float*)d_in[2];
  const float* sW2 = (const float*)d_in[3];
  const float* sb2 = (const float*)d_in[4];
  const float* sW3 = (const float*)d_in[5];
  const float* sb3 = (const float*)d_in[6];
  const float* eW1 = (const float*)d_in[7];
  const float* eb1 = (const float*)d_in[8];
  const float* eW2 = (const float*)d_in[9];
  const float* eb2 = (const float*)d_in[10];
  const float* eW3 = (const float*)d_in[11];
  const float* eb3 = (const float*)d_in[12];

  lila_prep<<<1, 256, 0, stream>>>(sW1, sb1, sW2, sb2, sW3, sb3,
                                   eW1, eb1, eW2, eb2, eW3, eb3, (char*)d_ws);
  lila_main<<<512, 256, 0, stream>>>(inp, (const char*)d_ws, (float*)d_out);
}
